// UniRNNCellL1_69191923139142
// MI455X (gfx1250) — matrix operand order repaired and checked by reading, not compiled, not run on hardware
//
#include <hip/hip_runtime.h>
#include <math.h>

typedef __attribute__((ext_vector_type(16))) _Float16 v16h;
typedef __attribute__((ext_vector_type(8)))  float    v8f;
typedef __attribute__((ext_vector_type(4)))  float    f4;

#define B_ 128
#define S_ 2048
#define I_ 128
#define H_ 128
#define C_ 10

// ---------------- workspace layout (bytes) ----------------
// [0      , 32768)   : wihB  : 8 ntiles x 4 kblocks x 32 lanes x 16 halves (B-frag layout)
// [32768  , 65536)   : whhB  : same layout
// [65536  , 66048)   : bias  : 128 floats (b_ih + b_hh)
// [66048  , 131584)  : lasth : 128 x 128 floats
// [262144 , +128MB+64KB) : xproj : [S+1][B][H] f32 (optional; +1 step pad for branchless prefetch)

// A-fragment (16x32 f16) position of element (row M, k-within-block Kk):
__device__ __forceinline__ int afrag_lane(int M, int Kk) {
    return M + (((Kk & 15) >= 8) ? 16 : 0);
}
__device__ __forceinline__ int afrag_j(int Kk) {
    int kk3 = Kk & 7;
    int v   = (kk3 >> 1) + ((Kk & 16) ? 4 : 0);
    return 2 * v + (kk3 & 1);
}

// branch-free tanh: tanh(x) = 1 - 2/(exp(2x)+1); saturates correctly at +/-inf.
__device__ __forceinline__ float tanh_fast(float x) {
    float e = __builtin_amdgcn_exp2f(x * 2.8853900817779268f);  // exp(2x)
    return 1.0f - 2.0f * __builtin_amdgcn_rcpf(e + 1.0f);
}

// ---------------- weight pack kernel ----------------
__global__ void rnn_pack_kernel(const float* __restrict__ W_ih, const float* __restrict__ b_ih,
                                const float* __restrict__ W_hh, const float* __restrict__ b_hh,
                                _Float16* __restrict__ wihB, _Float16* __restrict__ whhB,
                                float* __restrict__ bias) {
    const int tid = threadIdx.x;
    for (int idx = tid; idx < 8 * 4 * 32 * 16; idx += 256) {
        int j  = idx & 15;
        int L  = (idx >> 4) & 31;
        int kb = (idx >> 9) & 3;
        int nt = idx >> 11;
        int n  = nt * 16 + (L & 15);
        int k  = kb * 32 + ((j >> 3) << 4) + ((L >> 4) << 3) + (j & 7);
        wihB[idx] = (_Float16)W_ih[n * I_ + k];
        whhB[idx] = (_Float16)W_hh[n * H_ + k];
    }
    for (int n = tid; n < H_; n += 256) bias[n] = b_ih[n] + b_hh[n];
}

// ---------------- xproj precompute: xproj[t][b][n] = x[b,t,:]@W_ih^T + bias ----------------
// Fully parallel: B*(S/16) = 16384 workgroups, 8 waves each (wave w = n-tile w).
__launch_bounds__(256, 1)
__global__ void rnn_xproj_kernel(const float* __restrict__ x, const _Float16* __restrict__ wihB,
                                 const float* __restrict__ bias, float* __restrict__ xproj) {
    __shared__ alignas(32) _Float16 xs[4][32][16];
    const int tid  = threadIdx.x;
    const int lane = tid & 31;
    const int w    = tid >> 5;
    const int bb   = blockIdx.x / (S_ / 16);
    const int t0   = (blockIdx.x % (S_ / 16)) * 16;

    // stage 16 time-rows x[bb, t0..t0+15, :] into A-frag layout (one ds_store_b128/thread)
    const int r = tid >> 4, cb = tid & 15;
    const float* src = x + ((size_t)bb * S_ + t0 + r) * I_ + cb * 8;
    f4 p0 = *(const f4*)(src);
    f4 p1 = *(const f4*)(src + 4);
#pragma unroll
    for (int c = 0; c < 8; ++c) {
        float val = (c < 4) ? p0[c] : p1[c - 4];
        int K = cb * 8 + c;
        int kb = K >> 5, Kk = K & 31;
        xs[kb][afrag_lane(r, Kk)][afrag_j(Kk)] = (_Float16)val;
    }

    v16h wx[4];
#pragma unroll
    for (int kb = 0; kb < 4; ++kb)
        wx[kb] = *(const v16h*)(wihB + (((w * 4 + kb) * 32 + lane) << 4));
    const float bn = bias[w * 16 + (lane & 15)];
    __syncthreads();

    v8f c8;
#pragma unroll
    for (int v = 0; v < 8; ++v) c8[v] = bn;
#pragma unroll
    for (int kb = 0; kb < 4; ++kb) {
        v16h aX = *(const v16h*)(&xs[kb][lane][0]);
        c8 = __builtin_amdgcn_wmma_f32_16x16x32_f16(false, aX, false, wx[kb],
                                                    (short)0, c8, false, false);
    }

    // stores: single base address, immediate offsets v*B*H*4 = v*64KB (fits 24-bit ioffset)
    const int n = w * 16 + (lane & 15);
    const int Mbase = (lane >> 4) << 3;
    float* dst = xproj + ((size_t)(t0 + Mbase) * B_ + bb) * H_ + n;
#pragma unroll
    for (int v = 0; v < 8; ++v) dst[(size_t)v * B_ * H_] = c8[v];
}

// ---------------- lean recurrent scan (xproj precomputed) ----------------
// Per step: barrier -> 8 ds_load_b128 (h frags) -> 2 independent chains of 2 WMMAs
//           -> add register-prefetched xproj -> fast tanh -> scatter h -> capture last.
__launch_bounds__(256, 1)
__global__ void rnn_scan_xp_kernel(const float* __restrict__ xproj, const int* __restrict__ lens,
                                   const _Float16* __restrict__ whhB, float* __restrict__ lasth) {
    __shared__ alignas(32) _Float16 ha[2][4][32][16];

    const int tid  = threadIdx.x;
    const int lane = tid & 31;
    const int w    = tid >> 5;
    const int b0   = blockIdx.x * 16;

    for (int i = tid; i < 4 * 32 * 16; i += 256) (&ha[0][0][0][0])[i] = (_Float16)0.0f;

    v16h wh[4];
#pragma unroll
    for (int kb = 0; kb < 4; ++kb)
        wh[kb] = *(const v16h*)(whhB + (((w * 4 + kb) * 32 + lane) << 4));

    const int N     = lane & 15;
    const int n     = w * 16 + N;
    const int Mbase = (lane >> 4) << 3;
    const int kbh   = n >> 5, Kkh = n & 31;
    const int jh    = afrag_j(Kkh);
    const int hoff  = ((Kkh & 15) >= 8) ? 16 : 0;   // afrag_lane(v+Mbase,Kkh) = v+Mbase+hoff

    // loop-invariant store bases: scatter targets are base + v*16 halves (imm offsets)
    _Float16* hst[2] = { &ha[0][kbh][Mbase + hoff][jh], &ha[1][kbh][Mbase + hoff][jh] };
    float*    lastp  = lasth + (size_t)(b0 + Mbase) * H_ + n;  // + v*H_ (imm offsets)

    int lenm[8];
#pragma unroll
    for (int v = 0; v < 8; ++v) lenm[v] = lens[b0 + v + Mbase];

    // per-lane xproj pointer in C-fragment order; advance by B*H floats per step
    const float* xpp = xproj + (size_t)(b0 + Mbase) * H_ + n;
    float xp[8];
#pragma unroll
    for (int v = 0; v < 8; ++v) xp[v] = xpp[v * H_];  // t = 0
    xpp += (size_t)B_ * H_;                            // points at t = 1

    for (int t = 0; t < S_; ++t) {
        const int cur = t & 1, nxt = cur ^ 1;
        __syncthreads();  // ha[cur] ready (written as nxt last iter); WAR-safe by chaining

        v16h aH[4];
#pragma unroll
        for (int kb = 0; kb < 4; ++kb)
            aH[kb] = *(const v16h*)(&ha[cur][kb][lane][0]);

        // branchless prefetch of next step's xproj (xproj padded by one step)
        float xpn[8];
#pragma unroll
        for (int v = 0; v < 8; ++v) xpn[v] = xpp[v * H_];
        xpp += (size_t)B_ * H_;

        // two independent 2-deep WMMA chains over K = 128
        v8f c0, c1;
#pragma unroll
        for (int v = 0; v < 8; ++v) { c0[v] = 0.0f; c1[v] = 0.0f; }
        c0 = __builtin_amdgcn_wmma_f32_16x16x32_f16(false, aH[0], false, wh[0], (short)0, c0, false, false);
        c1 = __builtin_amdgcn_wmma_f32_16x16x32_f16(false, aH[1], false, wh[1], (short)0, c1, false, false);
        c0 = __builtin_amdgcn_wmma_f32_16x16x32_f16(false, aH[2], false, wh[2], (short)0, c0, false, false);
        c1 = __builtin_amdgcn_wmma_f32_16x16x32_f16(false, aH[3], false, wh[3], (short)0, c1, false, false);

        _Float16* hs = hst[nxt];
#pragma unroll
        for (int v = 0; v < 8; ++v) {
            float hv = tanh_fast(xp[v] + (c0[v] + c1[v]));
            hs[v * 16] = (_Float16)hv;                       // ds_store_b16, imm offset v*32B
            if (t + 1 == lenm[v]) lastp[v * H_] = hv;        // rare, execz-skipped
        }
#pragma unroll
        for (int v = 0; v < 8; ++v) xp[v] = xpn[v];
    }
}

// ---------------- fallback: fused scan (if ws too small for xproj) ----------------
__launch_bounds__(256, 1)
__global__ void rnn_scan_fused_kernel(const float* __restrict__ x, const int* __restrict__ lens,
                                      const _Float16* __restrict__ wihB, const _Float16* __restrict__ whhB,
                                      const float* __restrict__ bias, float* __restrict__ lasth) {
    __shared__ alignas(32) _Float16 xa[2][4][32][16];
    __shared__ alignas(32) _Float16 ha[2][4][32][16];

    const int tid  = threadIdx.x;
    const int lane = tid & 31;
    const int w    = tid >> 5;
    const int b0   = blockIdx.x * 16;

    for (int i = tid; i < 4 * 32 * 16; i += 256) (&ha[0][0][0][0])[i] = (_Float16)0.0f;

    v16h wx[4], wh[4];
#pragma unroll
    for (int kb = 0; kb < 4; ++kb) {
        wx[kb] = *(const v16h*)(wihB + (((w * 4 + kb) * 32 + lane) << 4));
        wh[kb] = *(const v16h*)(whhB + (((w * 4 + kb) * 32 + lane) << 4));
    }
    const float bn = bias[w * 16 + (lane & 15)];

    int lenm[8];
#pragma unroll
    for (int v = 0; v < 8; ++v) lenm[v] = lens[b0 + v + ((lane >> 4) << 3)];

    const int r  = tid >> 4;
    const int cb = tid & 15;
    const float* xrow = x + (size_t)(b0 + r) * S_ * I_ + cb * 8;
    f4 p0 = *(const f4*)(xrow);
    f4 p1 = *(const f4*)(xrow + 4);

    for (int t = 0; t < S_; ++t) {
        const int cur = t & 1, nxt = cur ^ 1;
#pragma unroll
        for (int c = 0; c < 8; ++c) {
            float val = (c < 4) ? p0[c] : p1[c - 4];
            int K = cb * 8 + c;
            int kb = K >> 5, Kk = K & 31;
            xa[cur][kb][afrag_lane(r, Kk)][afrag_j(Kk)] = (_Float16)val;
        }
        if (t + 1 < S_) {
            const float* nx = xrow + (size_t)(t + 1) * I_;
            p0 = *(const f4*)(nx);
            p1 = *(const f4*)(nx + 4);
        }
        __syncthreads();

        v16h aX[4], aH[4];
#pragma unroll
        for (int kb = 0; kb < 4; ++kb) {
            aX[kb] = *(const v16h*)(&xa[cur][kb][lane][0]);
            aH[kb] = *(const v16h*)(&ha[cur][kb][lane][0]);
        }
        v8f cX, cH;
#pragma unroll
        for (int v = 0; v < 8; ++v) { cX[v] = bn; cH[v] = 0.0f; }
#pragma unroll
        for (int kb = 0; kb < 4; ++kb) {
            cX = __builtin_amdgcn_wmma_f32_16x16x32_f16(false, aX[kb], false, wx[kb], (short)0, cX, false, false);
            cH = __builtin_amdgcn_wmma_f32_16x16x32_f16(false, aH[kb], false, wh[kb], (short)0, cH, false, false);
        }

        const int N   = lane & 15;
        const int n   = w * 16 + N;
        const int kbh = n >> 5, Kkh = n & 31;
        const int jh  = afrag_j(Kkh);
#pragma unroll
        for (int v = 0; v < 8; ++v) {
            int   M  = v + ((lane >> 4) << 3);
            float hv = tanh_fast(cX[v] + cH[v]);
            ha[nxt][kbh][afrag_lane(M, Kkh)][jh] = (_Float16)hv;
            if (t + 1 == lenm[v]) lasth[(size_t)(b0 + M) * H_ + n] = hv;
        }
    }
}

// ---------------- FC + softmax ----------------
__global__ void rnn_fc_softmax_kernel(const float* __restrict__ lasth, const float* __restrict__ W_fc,
                                      const float* __restrict__ b_fc, float* __restrict__ out) {
    int b = blockIdx.x * blockDim.x + threadIdx.x;
    if (b >= B_) return;
    const float* hrow = lasth + (size_t)b * H_;
    float logits[C_];
    float mx = -1e30f;
#pragma unroll
    for (int c = 0; c < C_; ++c) {
        float acc = b_fc[c];
        for (int k = 0; k < H_; ++k) acc += hrow[k] * W_fc[c * H_ + k];
        logits[c] = acc;
        mx = fmaxf(mx, acc);
    }
    float sum = 0.0f;
#pragma unroll
    for (int c = 0; c < C_; ++c) {
        logits[c] = __builtin_amdgcn_exp2f((logits[c] - mx) * 1.4426950408889634f);
        sum += logits[c];
    }
    float inv = 1.0f / sum;
#pragma unroll
    for (int c = 0; c < C_; ++c) out[b * C_ + c] = logits[c] * inv;
}

extern "C" void kernel_launch(void* const* d_in, const int* in_sizes, int n_in,
                              void* d_out, int out_size, void* d_ws, size_t ws_size,
                              hipStream_t stream) {
    const float* x         = (const float*)d_in[0];
    const int*   X_lengths = (const int*)d_in[1];
    const float* W_ih      = (const float*)d_in[2];
    const float* b_ih      = (const float*)d_in[3];
    const float* W_hh      = (const float*)d_in[4];
    const float* b_hh      = (const float*)d_in[5];
    const float* W_fc      = (const float*)d_in[6];
    const float* b_fc      = (const float*)d_in[7];
    float*       out       = (float*)d_out;

    char* ws = (char*)d_ws;
    _Float16* wihB  = (_Float16*)(ws + 0);
    _Float16* whhB  = (_Float16*)(ws + 32768);
    float*    bias  = (float*)(ws + 65536);
    float*    lasth = (float*)(ws + 66048);
    float*    xproj = (float*)(ws + 262144);

    // +1 time-step of padding so the scan's prefetch is branchless
    const size_t need_big = 262144 + (size_t)(S_ + 1) * B_ * H_ * sizeof(float);

    rnn_pack_kernel<<<1, 256, 0, stream>>>(W_ih, b_ih, W_hh, b_hh, wihB, whhB, bias);
    if (ws_size >= need_big) {
        rnn_xproj_kernel<<<B_ * (S_ / 16), 256, 0, stream>>>(x, wihB, bias, xproj);
        rnn_scan_xp_kernel<<<B_ / 16, 256, 0, stream>>>(xproj, X_lengths, whhB, lasth);
    } else {
        rnn_scan_fused_kernel<<<B_ / 16, 256, 0, stream>>>(x, X_lengths, wihB, whhB, bias, lasth);
    }
    rnn_fc_softmax_kernel<<<1, 128, 0, stream>>>(lasth, W_fc, b_fc, out);
}
